// Model_17789754540549
// MI455X (gfx1250) — matrix operand order repaired and checked by reading, not compiled, not run on hardware
//
#include <hip/hip_runtime.h>
#include <stdint.h>

// ---------------------------------------------------------------------------
// Fused attention-like kernel for gfx1250 (MI455X):
//   out = dropout(softmax(25 * x x^T)) @ x ,  x : [8192,1024] f32
// bf16 WMMA (v_wmma_f32_16x16x32_bf16), two-pass softmax, O in registers.
// ---------------------------------------------------------------------------

typedef __attribute__((ext_vector_type(16))) __bf16 v16bf;
typedef __attribute__((ext_vector_type(8)))  __bf16 v8bf;
typedef __attribute__((ext_vector_type(8)))  float  v8f;

#define N_ROWS   8192
#define D_DIM    1024
#define QK_SCALE 25.0f
#define KEEP_P   0.8f
#define INV_KEEP 1.25f

#define BLK_M 32            // rows of the output handled per workgroup
#define BLK_K 64            // key columns processed per k-iteration
#define QSTR  1032          // q_lds row stride in elements (pad: 2064B = 516 dw, 516%64=4 -> no bank conflicts)
#define PSTR  72            // p_lds row stride in elements (144B = 36 dw, gcd(36,64)=4 -> 16 distinct banks)

// dynamic LDS layout (bytes)
#define SM_Q     0
#define SM_P     (BLK_M * QSTR * 2)                 // 66048
#define SM_STAT  (SM_P + 2 * BLK_M * PSTR * 2)      // + 9216  = 75264
#define SM_MROW  (SM_STAT + 8 * 16 * 2 * 4)         // + 1024  = 76288
#define SM_COEF  (SM_MROW + BLK_M * 4)              // + 128   = 76416
#define SM_TOTAL (SM_COEF + BLK_M * 4)              // 76544

// ---------------------------------------------------------------------------
// helpers
// ---------------------------------------------------------------------------

__device__ __forceinline__ uint16_t f2bf(float f) {
    uint32_t u = __float_as_uint(f);
    uint32_t r = u + 0x7fffu + ((u >> 16) & 1u);   // round-to-nearest-even
    return (uint16_t)(r >> 16);
}

// deterministic per-(i,j) dropout decision (counter-based hash -> U[0,1))
__device__ __forceinline__ bool keep_elt(uint32_t i, uint32_t j) {
    uint32_t s = i * 0x9E3779B1u + j * 0x85EBCA77u + 0xC2B2AE3Du;
    s ^= s >> 16; s *= 0x7FEB352Du;
    s ^= s >> 15; s *= 0x846CA68Bu;
    s ^= s >> 16;
    return (float)(s >> 8) * (1.0f / 16777216.0f) < KEEP_P;
}

__device__ __forceinline__ v8f wmma_bf16(v16bf a, v16bf b, v8f c) {
    return __builtin_amdgcn_wmma_f32_16x16x32_bf16(
        /*neg_a=*/false, a, /*neg_b=*/false, b,
        /*c_mod=*/(short)0, c, /*reuse_a=*/false, /*reuse_b=*/false);
}

// A-operand (16x32 bf16) load from a row-major tile.
// ISA layout: lane L holds row m=L&15, K = base..base+7 and base+16..base+23
// with base = (L>=16) ? 8 : 0  -> two contiguous 16B chunks per lane.
__device__ __forceinline__ v16bf load_a(const uint16_t* __restrict__ tile,
                                        int stride, int k0, int lane) {
    int m  = lane & 15;
    int kb = (lane & 16) ? 8 : 0;
    const uint16_t* p = tile + m * stride + k0 + kb;
    v8bf lo = *(const v8bf*)(p);
    v8bf hi = *(const v8bf*)(p + 16);
    return __builtin_shufflevector(lo, hi,
        0, 1, 2, 3, 4, 5, 6, 7, 8, 9, 10, 11, 12, 13, 14, 15);
}

// S = (Q K^T) tile for 16 rows x 16 cols, K-reduced over D=1024.
// B operand (32x16 bf16): lane L holds column n=L&15, K in the same order as A
// (base..base+7, base+16..base+23), gathered from row-major xT ([D][N] = K^T).
__device__ __forceinline__ void compute_s_tile(const uint16_t* __restrict__ qtile,
                                               const uint16_t* __restrict__ xT,
                                               int ncol0, int lane,
                                               float sv[8]) {
    v8f c0 = {};
    v8f c1 = {};
    for (int d0 = 0; d0 < D_DIM; d0 += 64) {
        v16bf a0 = load_a(qtile, QSTR, d0, lane);
        v16bf b0, b1;
#pragma unroll
        for (int s = 0; s < 16; ++s) {
            int kk = (s & 7) + ((lane & 16) ? 8 : 0) + ((s & 8) ? 16 : 0);
            b0[s] = ((const __bf16*)xT)[(size_t)(d0 + kk) * N_ROWS + ncol0 + (lane & 15)];
            b1[s] = ((const __bf16*)xT)[(size_t)(d0 + 32 + kk) * N_ROWS + ncol0 + (lane & 15)];
        }
        c0 = wmma_bf16(a0, b0, c0);
        v16bf a1 = load_a(qtile, QSTR, d0 + 32, lane);

        c1 = wmma_bf16(a1, b1, c1);
    }
#pragma unroll
    for (int r = 0; r < 8; ++r) sv[r] = (c0[r] + c1[r]) * QK_SCALE;
}

// ---------------------------------------------------------------------------
// prep: x f32 -> bf16 row-major (xw) and bf16 transposed (xT)
// ---------------------------------------------------------------------------
__global__ void prep_bf16(const float* __restrict__ x,
                          uint16_t* __restrict__ xw,
                          uint16_t* __restrict__ xT) {
    int idx = blockIdx.x * blockDim.x + threadIdx.x;
    int i = idx >> 10;        // row   (D_DIM = 1024)
    int j = idx & 1023;       // col
    uint16_t b = f2bf(x[idx]);
    xw[idx] = b;
    xT[(size_t)j * N_ROWS + i] = b;
}

// ---------------------------------------------------------------------------
// main fused kernel: one workgroup = 32 output rows, 8 waves.
//   wave wv: S-tile (rt = wv>>2 row half, tc = wv&3 col quarter),
//            PV d-slice = columns [wv*128, wv*128+128).
// ---------------------------------------------------------------------------
__global__ void __launch_bounds__(256)
attn_fused(const uint16_t* __restrict__ xw,
           const uint16_t* __restrict__ xT,
           float* __restrict__ out) {
    extern __shared__ char smem[];
    uint16_t* q    = (uint16_t*)(smem + SM_Q);
    uint16_t* pb   = (uint16_t*)(smem + SM_P);
    float*    stat = (float*)(smem + SM_STAT);   // [8 waves][16 rows][m,l]
    float*    mrow = (float*)(smem + SM_MROW);   // [32]
    float*    coef = (float*)(smem + SM_COEF);   // [32] = 1.25 / l

    const int lane = threadIdx.x & 31;
    const int wv   = threadIdx.x >> 5;
    const int rt   = wv >> 2;        // 0/1 : rows 0-15 or 16-31 of S block
    const int tc   = wv & 3;         // column quarter of the 64-wide k-block
    const int blockrow = blockIdx.x * BLK_M;
    const int hh8 = (lane & 16) ? 8 : 0;

    // ---- stage Q block (32 x 1024 bf16) into padded LDS ----
    for (int idx = threadIdx.x; idx < BLK_M * (D_DIM / 8); idx += blockDim.x) {
        int row = idx >> 7;          // 128 8-elem chunks per row
        int ch  = idx & 127;
        uint4 v = *(const uint4*)(xw + (size_t)(blockrow + row) * D_DIM + ch * 8);
        *(uint4*)(q + row * QSTR + ch * 8) = v;
    }
    __syncthreads();

    const uint16_t* qtile = q + rt * 16 * QSTR;

    // ================= pass 1: row max / sum-exp (online, no barriers) ======
    float m_run[8], l_run[8];
#pragma unroll
    for (int r = 0; r < 8; ++r) { m_run[r] = -3.0e38f; l_run[r] = 0.0f; }

    for (int kb = 0; kb < N_ROWS; kb += BLK_K) {
        float sv[8];
        compute_s_tile(qtile, xT, kb + tc * 16, lane, sv);
#pragma unroll
        for (int r = 0; r < 8; ++r) {
            float s  = sv[r];
            float bm = s;                                 // row max over 16 lanes
            bm = fmaxf(bm, __shfl_xor(bm, 1, 32));
            bm = fmaxf(bm, __shfl_xor(bm, 2, 32));
            bm = fmaxf(bm, __shfl_xor(bm, 4, 32));
            bm = fmaxf(bm, __shfl_xor(bm, 8, 32));
            float mnew = fmaxf(m_run[r], bm);
            float e = __expf(s - mnew);                   // row sum over 16 lanes
            e += __shfl_xor(e, 1, 32);
            e += __shfl_xor(e, 2, 32);
            e += __shfl_xor(e, 4, 32);
            e += __shfl_xor(e, 8, 32);
            l_run[r] = l_run[r] * __expf(m_run[r] - mnew) + e;
            m_run[r] = mnew;
        }
    }

    // merge the 4 column-quarter partials per row
    if ((lane & 15) == 0) {
        int hh = lane >> 4;
#pragma unroll
        for (int r = 0; r < 8; ++r) {
            int rl = hh * 8 + r;
            stat[(wv * 16 + rl) * 2 + 0] = m_run[r];
            stat[(wv * 16 + rl) * 2 + 1] = l_run[r];
        }
    }
    __syncthreads();
    if (threadIdx.x < BLK_M) {
        int row = threadIdx.x;
        int rg  = row >> 4;
        int rl  = row & 15;
        float M = -3.0e38f;
        for (int cg = 0; cg < 4; ++cg)
            M = fmaxf(M, stat[((rg * 4 + cg) * 16 + rl) * 2 + 0]);
        float L = 0.0f;
        for (int cg = 0; cg < 4; ++cg) {
            float mw = stat[((rg * 4 + cg) * 16 + rl) * 2 + 0];
            float lw = stat[((rg * 4 + cg) * 16 + rl) * 2 + 1];
            L += lw * __expf(mw - M);
        }
        mrow[row] = M;
        coef[row] = INV_KEEP / L;
    }
    __syncthreads();

    // ================= pass 2: P = dropout(softmax(S)), O += P @ V ==========
    v8f o[2][8];
#pragma unroll
    for (int i = 0; i < 2; ++i)
#pragma unroll
        for (int j = 0; j < 8; ++j) o[i][j] = (v8f){};

    const int col_loc = tc * 16 + (lane & 15);
    int it = 0;
    for (int kb = 0; kb < N_ROWS; kb += BLK_K, ++it) {
        uint16_t* pcur = pb + (it & 1) * (BLK_M * PSTR);

        // prefetch next k-block's V rows for this wave's d-slice
        if (kb + BLK_K < N_ROWS)
            __builtin_prefetch(xw + (size_t)(kb + BLK_K + lane) * D_DIM + wv * 128, 0, 1);

        float sv[8];
        compute_s_tile(qtile, xT, kb + tc * 16, lane, sv);

#pragma unroll
        for (int r = 0; r < 8; ++r) {
            int row_loc = rt * 16 + r + hh8;
            float pv = __expf(sv[r] - mrow[row_loc]) * coef[row_loc];
            uint32_t gi = (uint32_t)(blockrow + row_loc);
            uint32_t gj = (uint32_t)(kb + col_loc);
            if (!keep_elt(gi, gj)) pv = 0.0f;
            pcur[row_loc * PSTR + col_loc] = f2bf(pv);
        }
        __syncthreads();   // single barrier per iter (P double-buffered)

#pragma unroll
        for (int ks = 0; ks < 2; ++ks) {
            v16bf a0 = load_a(pcur,             PSTR, ks * 32, lane);
            v16bf a1 = load_a(pcur + 16 * PSTR, PSTR, ks * 32, lane);
#pragma unroll
            for (int dt = 0; dt < 8; ++dt) {
                v16bf b = load_a(xT + (size_t)(wv * 128 + dt * 16) * N_ROWS,
                    N_ROWS, kb + ks * 32, lane);
                o[0][dt] = wmma_bf16(a0, b, o[0][dt]);
                o[1][dt] = wmma_bf16(a1, b, o[1][dt]);
            }
        }
    }

    // ---- epilogue: write O (f32) ----
#pragma unroll
    for (int rg = 0; rg < 2; ++rg)
#pragma unroll
        for (int dt = 0; dt < 8; ++dt)
#pragma unroll
            for (int r = 0; r < 8; ++r) {
                int row = blockrow + rg * 16 + r + hh8;
                int col = wv * 128 + dt * 16 + (lane & 15);
                out[(size_t)row * D_DIM + col] = o[rg][dt][r];
            }
}

// ---------------------------------------------------------------------------
extern "C" void kernel_launch(void* const* d_in, const int* in_sizes, int n_in,
                              void* d_out, int out_size, void* d_ws, size_t ws_size,
                              hipStream_t stream) {
    (void)in_sizes; (void)n_in; (void)out_size; (void)ws_size;
    const float* x = (const float*)d_in[0];
    float* out = (float*)d_out;
    uint16_t* xw = (uint16_t*)d_ws;                       // 16 MB bf16 row-major
    uint16_t* xT = xw + (size_t)N_ROWS * D_DIM;           // 16 MB bf16 transposed

    prep_bf16<<<(N_ROWS * D_DIM) / 256, 256, 0, stream>>>(x, xw, xT);
    attn_fused<<<N_ROWS / BLK_M, 256, SM_TOTAL, stream>>>(xw, xT, out);
}
